// NodeAttentionLayer_32744830664926
// MI455X (gfx1250) — matrix operand order repaired and checked by reading, not compiled, not run on hardware
//
#include <hip/hip_runtime.h>

typedef __attribute__((ext_vector_type(16))) __bf16 v16bf;
typedef __attribute__((ext_vector_type(8)))  float  v8f;
typedef __attribute__((__vector_size__(16))) int    v4i;

#define N_NODES 8192
#define F_IN    256
#define F_OUT   64
#define ALPHA   0.2f
#define NEG_BIG -9.0e15f
#define M_INIT  -3.0e38f
#define JT      64          // j-tile for fused attention

#define AS1 __attribute__((address_space(1)))
#define AS3 __attribute__((address_space(3)))

#if __has_builtin(__builtin_amdgcn_global_load_async_to_lds_b128) && \
    __has_builtin(__builtin_amdgcn_s_wait_asynccnt)
#define ASYNC_COPY 1
#else
#define ASYNC_COPY 0
#endif

// 16-byte global->LDS copy: async-to-LDS on gfx1250 toolchains that expose it,
// plain VGPR bounce otherwise.
static __device__ __forceinline__ void cp16(void* dst_lds, const void* src_gbl) {
#if ASYNC_COPY
  __builtin_amdgcn_global_load_async_to_lds_b128(
      (AS1 v4i*)(v4i*)src_gbl, (AS3 v4i*)(v4i*)dst_lds, 0, 0);
#else
  *reinterpret_cast<uint4*>(dst_lds) = *reinterpret_cast<const uint4*>(src_gbl);
#endif
}

// async instructions issued per wave per tile stage: 4 (hoT) + 8 (adj) + 1 (so)
#define ASYNC_PER_TILE 13

// ---------------------------------------------------------------------------
// Kernel 0: fold attention vector into weights: v_t = W_t @ a1, v_o = W_o @ a2
// ---------------------------------------------------------------------------
__global__ __launch_bounds__(256) void gat_foldvec(
    const float* __restrict__ Wt, const float* __restrict__ Wo,
    const float* __restrict__ a, float* __restrict__ vt, float* __restrict__ vo) {
  int k = threadIdx.x;
  float s1 = 0.f, s2 = 0.f;
#pragma unroll 8
  for (int f = 0; f < F_OUT; ++f) {
    s1 += Wt[k * F_OUT + f] * a[f];
    s2 += Wo[k * F_OUT + f] * a[F_OUT + f];
  }
  vt[k] = s1;
  vo[k] = s2;
}

// ---------------------------------------------------------------------------
// Kernel 1: s_t = t_input @ v_t ; s_o = o_input @ v_o (memory-bound matvec)
// ---------------------------------------------------------------------------
__global__ __launch_bounds__(256) void gat_scores(
    const float* __restrict__ t_in, const float* __restrict__ o_in,
    const float* __restrict__ vt, const float* __restrict__ vo,
    float* __restrict__ st, float* __restrict__ so) {
  const float* X = (blockIdx.y == 0) ? t_in : o_in;
  const float* V = (blockIdx.y == 0) ? vt : vo;
  float* S       = (blockIdx.y == 0) ? st : so;
  int wv   = threadIdx.x >> 5;
  int lane = threadIdx.x & 31;
  int row  = blockIdx.x * 8 + wv;
  float sum = 0.f;
#pragma unroll
  for (int i = 0; i < F_IN / 32; ++i) {
    int c = lane + 32 * i;
    sum += X[(size_t)row * F_IN + c] * V[c];
  }
#pragma unroll
  for (int off = 16; off > 0; off >>= 1) sum += __shfl_xor(sum, off);
  if (lane == 0) S[row] = sum;
}

// ---------------------------------------------------------------------------
// Kernel 2: hoT = (o_input @ W_o)^T as bf16 [64 x 8192], computed directly in
// transposed form (A = W_o^T staged transposed in LDS, B = o_input tile).
// Both LDS operand reads are contiguous (ds_load_b128) and the global b16
// stores are lane-coalesced.
// ---------------------------------------------------------------------------
__global__ __launch_bounds__(256) void gat_proj_hoT(
    const float* __restrict__ o_in, const float* __restrict__ Wo,
    __bf16* __restrict__ hoT) {
  const int lane = threadIdx.x & 31;
  const int wv   = threadIdx.x >> 5;   // 0..7
  const int hi   = lane >> 4;
  const int mrow = lane & 15;
  const int nbase = blockIdx.x * 128;  // block covers 128 nodes
  const int nwave = nbase + wv * 16;   // this wave's 16 nodes

  __shared__ __align__(16) __bf16 sh_wt[F_OUT * F_IN];  // W_o^T [f][k], 32 KB
  __shared__ __align__(16) __bf16 sh_x[128 * 32];       // o_in tile [node][k], 8 KB

  // stage W_o transposed (one-time scattered LDS stores, coalesced global reads)
#pragma unroll
  for (int i = 0; i < (F_IN * F_OUT) / 256; ++i) {
    int idx = threadIdx.x + 256 * i;
    int k = idx >> 6, f = idx & 63;
    sh_wt[f * F_IN + k] = (__bf16)Wo[idx];
  }

  v8f acc[4] = {v8f{}, v8f{}, v8f{}, v8f{}};

  for (int k0 = 0; k0 < F_IN; k0 += 32) {
    __syncthreads();
#pragma unroll
    for (int i = 0; i < (128 * 32) / 256; ++i) {
      int idx = threadIdx.x + 256 * i;
      int r = idx >> 5, c = idx & 31;
      sh_x[idx] = (__bf16)o_in[(size_t)(nbase + r) * F_IN + k0 + c];
    }
    __syncthreads();

    // B: 32x16 (K x nodes): lane -> node nwave+mrow, K = i + 16*hi (contiguous)
    v16bf bv;
#pragma unroll
    for (int i = 0; i < 16; ++i)
      bv[i] = sh_x[(wv * 16 + mrow) * 32 + i + 16 * hi];

    // A: 16x32 (f x K) per f-tile: lane -> f row mrow, K = i + 16*hi (contiguous)
#pragma unroll
    for (int ft = 0; ft < 4; ++ft) {
      v16bf av;
#pragma unroll
      for (int i = 0; i < 16; ++i)
        av[i] = sh_wt[(ft * 16 + mrow) * F_IN + k0 + i + 16 * hi];
      acc[ft] = __builtin_amdgcn_wmma_f32_16x16x32_bf16(
          false, av, false, bv, (short)0, acc[ft], false, false);
    }
  }

  // C: row = f (v + 8*hi within f-tile), col = node (lane%16) -> coalesced b16
#pragma unroll
  for (int v = 0; v < 8; ++v)
#pragma unroll
    for (int ft = 0; ft < 4; ++ft)
      hoT[(size_t)(ft * 16 + v + 8 * hi) * N_NODES + nwave + mrow] =
          (__bf16)acc[ft][v];
}

// ---------------------------------------------------------------------------
// Kernel 3: fused masked-softmax attention (flash style) + ELU.
// Double-buffered async-to-LDS staging of adj / hoT / s_o tiles; online
// softmax in registers feeds bf16 WMMA A; B read contiguously from hoT tile.
// ---------------------------------------------------------------------------
__global__ __launch_bounds__(128) void gat_attn(
    const float* __restrict__ st, const float* __restrict__ so,
    const int* __restrict__ adj, const __bf16* __restrict__ hoT,
    float* __restrict__ out) {
  const int lane = threadIdx.x & 31;
  const int wv   = threadIdx.x >> 5;   // 0..3
  const int hi   = lane >> 4;
  const int mrow = lane & 15;
  const int rowbase = blockIdx.x * 64;
  const int wrow    = rowbase + wv * 16;

  __shared__ __align__(16) __bf16 sh_hoT[2][JT * F_OUT];  // [f][j], 8 KB each
  __shared__ __align__(16) int    sh_adj[2][64 * JT];     // [row][j], 16 KB each
  __shared__ __align__(16) float  sh_so[2][JT];

  v8f acc[4] = {v8f{}, v8f{}, v8f{}, v8f{}};
  float m = M_INIT, lsum = 0.f;
  const float strow = st[wrow + mrow];

  // stage one j-tile into LDS buffer `buf` (13 async instrs per wave)
  auto stage = [&](int buf, int j0) {
    if (lane < 4)  // each wave covers its own 16-float slice of s_o
      cp16(&sh_so[buf][16 * wv + 4 * lane], &so[j0 + 16 * wv + 4 * lane]);
#pragma unroll
    for (int i = 0; i < 4; ++i) {     // hoT tile: 64f x 64j bf16 = 512 x 16B
      int idx = threadIdx.x + 128 * i;
      int f = idx >> 3, c = idx & 7;
      cp16(&sh_hoT[buf][f * JT + c * 8],
           &hoT[(size_t)f * N_NODES + j0 + c * 8]);
    }
#pragma unroll
    for (int i = 0; i < 8; ++i) {     // adj tile: 64r x 64j int32 = 1024 x 16B
      int idx = threadIdx.x + 128 * i;
      int r = idx >> 4, c = idx & 15;
      cp16(&sh_adj[buf][r * JT + c * 4],
           &adj[(size_t)(rowbase + r) * N_NODES + j0 + c * 4]);
    }
  };

  constexpr int NTILES = N_NODES / JT;
  stage(0, 0);

  for (int t = 0; t < NTILES; ++t) {
    const int cur = t & 1;
    const bool has_next = (t + 1 < NTILES);
    if (has_next) stage(1 - cur, (t + 1) * JT);
#if ASYNC_COPY
    if (has_next) __builtin_amdgcn_s_wait_asynccnt(ASYNC_PER_TILE);
    else          __builtin_amdgcn_s_wait_asynccnt(0);
#endif
    __syncthreads();  // tile `cur` visible to all waves

#pragma unroll
    for (int ks = 0; ks < 2; ++ks) {
      const int jb = ks * 32;
      // e_ij = leakyrelu(s_t+s_o) masked by adj, laid out per WMMA A map
      float ev[16];
      float tmax = M_INIT;
#pragma unroll
      for (int i = 0; i < 16; ++i) {
        int koff = i + 16 * hi;
        int jl = jb + koff;
        float e = strow + sh_so[cur][jl];
        e = (e > 0.f) ? e : ALPHA * e;
        e = (sh_adj[cur][(wv * 16 + mrow) * JT + jl] > 0) ? e : NEG_BIG;
        ev[i] = e;
        tmax = fmaxf(tmax, e);
      }
      tmax = fmaxf(tmax, __shfl_xor(tmax, 16));
      float newm = fmaxf(m, tmax);
      float corr = __expf(m - newm);
      m = newm;

      v16bf av;
      float rs = 0.f;
#pragma unroll
      for (int i = 0; i < 16; ++i) {
        float p = __expf(ev[i] - m);
        rs += p;
        av[i] = (__bf16)p;
      }
      rs += __shfl_xor(rs, 16);
      lsum = lsum * corr + rs;

      // rescale accumulators (C row = v + 8*hi)
#pragma unroll
      for (int v = 0; v < 8; ++v) {
        float cf = __shfl(corr, v + 8 * hi);
        acc[0][v] *= cf; acc[1][v] *= cf; acc[2][v] *= cf; acc[3][v] *= cf;
      }

      // acc += P @ h_o ; B from transposed tile -> contiguous ds_load_b128
#pragma unroll
      for (int nt = 0; nt < 4; ++nt) {
        v16bf bv;
        const int base = (mrow + 16 * nt) * JT + jb + 16 * hi;
#pragma unroll
        for (int i = 0; i < 16; ++i) bv[i] = sh_hoT[cur][base + i];
        acc[nt] = __builtin_amdgcn_wmma_f32_16x16x32_bf16(
            false, av, false, bv, (short)0, acc[nt], false, false);
      }
    }
    __syncthreads();  // done reading buffer `cur`; safe to overwrite at t+2
  }

  // epilogue: normalize, ELU, store f32
  float linv = 1.f / lsum;
#pragma unroll
  for (int v = 0; v < 8; ++v) {
    float li = __shfl(linv, v + 8 * hi);
    int grow = wrow + v + 8 * hi;
#pragma unroll
    for (int nt = 0; nt < 4; ++nt) {
      float x = acc[nt][v] * li;
      x = (x > 0.f) ? x : (__expf(x) - 1.f);
      out[(size_t)grow * F_OUT + mrow + 16 * nt] = x;
    }
  }
}

// ---------------------------------------------------------------------------
extern "C" void kernel_launch(void* const* d_in, const int* in_sizes, int n_in,
                              void* d_out, int out_size, void* d_ws, size_t ws_size,
                              hipStream_t stream) {
  const float* t_input = (const float*)d_in[0];
  const float* o_input = (const float*)d_in[1];
  const int*   adj     = (const int*)d_in[2];
  const float* Wt      = (const float*)d_in[3];
  const float* Wo      = (const float*)d_in[4];
  const float* a       = (const float*)d_in[5];
  float* out = (float*)d_out;

  float* wsf = (float*)d_ws;
  float* vt = wsf;           // 256
  float* vo = wsf + 256;     // 256
  float* st = wsf + 512;     // 8192
  float* so = wsf + 8704;    // 8192
  __bf16* hoT = (__bf16*)((char*)d_ws + 16896 * sizeof(float));  // [64][8192] bf16

  gat_foldvec<<<1, 256, 0, stream>>>(Wt, Wo, a, vt, vo);
  gat_scores<<<dim3(N_NODES / 8, 2), 256, 0, stream>>>(t_input, o_input, vt, vo, st, so);
  gat_proj_hoT<<<N_NODES / 128, 256, 0, stream>>>(o_input, Wo, hoT);
  gat_attn<<<N_NODES / 64, 128, 0, stream>>>(st, so, adj, hoT, out);
}
